// Mamba_7945689497800
// MI455X (gfx1250) — matrix operand order repaired and checked by reading, not compiled, not run on hardware
//
#include <hip/hip_runtime.h>
#include <hip/hip_bf16.h>

#define NLAYER 4
#define DMODEL 1024
#define DINNER 2048
#define DSTATE 16
#define DTRANK 64
#define DCONV  4
#define BATCH  2
#define SEQLEN 2048
#define MROWS  (BATCH * SEQLEN)        // 4096 token rows (always %128 == 0)
#define XDBLW  (DTRANK + 2 * DSTATE)   // 96

typedef __attribute__((ext_vector_type(16))) __bf16 v16bf;
typedef __attribute__((ext_vector_type(8)))  __bf16 v8bf;
typedef __attribute__((ext_vector_type(8)))  float  v8f;

// ---------------------------------------------------------------------------
// fp32 -> bf16 conversion (grid-stride)
// ---------------------------------------------------------------------------
__global__ void k_cvt_bf16(const float* __restrict__ in, __bf16* __restrict__ out, long n) {
    long i = (long)blockIdx.x * blockDim.x + threadIdx.x;
    long st = (long)gridDim.x * blockDim.x;
    for (; i < n; i += st) out[i] = (__bf16)in[i];
}

// ---------------------------------------------------------------------------
// residual add + LayerNorm over DMODEL; optional bf16 and/or f32 output
// one block (256 thr) per token row of 1024
// ---------------------------------------------------------------------------
__global__ void k_add_ln(float* __restrict__ resid, const float* __restrict__ add,
                         const float* __restrict__ w, const float* __restrict__ b,
                         __bf16* out_bf, float* out_f) {
    __shared__ float red[256];
    int row = blockIdx.x;
    size_t base = (size_t)row * DMODEL;
    float v[4];
    float s = 0.f;
#pragma unroll
    for (int i = 0; i < 4; ++i) {
        int c = threadIdx.x + 256 * i;
        float x = resid[base + c];
        if (add) { x += add[base + c]; resid[base + c] = x; }
        v[i] = x; s += x;
    }
    red[threadIdx.x] = s; __syncthreads();
    for (int o = 128; o > 0; o >>= 1) {
        if (threadIdx.x < o) red[threadIdx.x] += red[threadIdx.x + o];
        __syncthreads();
    }
    float mu = red[0] * (1.f / DMODEL);
    __syncthreads();
    float s2 = 0.f;
#pragma unroll
    for (int i = 0; i < 4; ++i) { float t = v[i] - mu; s2 += t * t; }
    red[threadIdx.x] = s2; __syncthreads();
    for (int o = 128; o > 0; o >>= 1) {
        if (threadIdx.x < o) red[threadIdx.x] += red[threadIdx.x + o];
        __syncthreads();
    }
    float rstd = rsqrtf(red[0] * (1.f / DMODEL) + 1e-5f);
#pragma unroll
    for (int i = 0; i < 4; ++i) {
        int c = threadIdx.x + 256 * i;
        float outv = (v[i] - mu) * rstd * w[c] + b[c];
        if (out_bf) out_bf[base + c] = (__bf16)outv;
        if (out_f)  out_f[base + c]  = outv;
    }
}

// ---------------------------------------------------------------------------
// WMMA bf16 GEMM, NT layout: C[M,N] = A[M,K] * W[N,K]^T, f32 accumulate
// block = 256 thr = 8 waves, arranged 4 (M) x 2 (N); wave tile = 32x64
// workgroup tile = 128x128; K stepped by 32.
// B tiles (128 rows x 64B) staged global->LDS with async copies, double
// buffered. Last K-iteration is peeled so the steady-state loop is branch-
// free: issue tile k+1, s_wait_asynccnt 2 (tile k done, in-order), barrier,
// 8 WMMAs from LDS, barrier.
//
// Fragment layouts (A/B k order is this code's contiguous choice, not the ISA table; C/D per ISA):
//  A 16x32 bf16: lane m(0..15) holds K 0-15 contiguous; lane m+16 holds K 16-31
//  B 32x16 bf16: lane n holds K 0-15 contiguous; lane n+16 holds K 16-31
//  C/D 16x16 f32: lane = {mhalf,n}; VGPR r -> row mhalf*8+r
// ---------------------------------------------------------------------------
__device__ inline v16bf load_a_frag(const __bf16* __restrict__ A, int lda,
                                    int mBase, int k0, int lane) {
    int m = mBase + (lane & 15);
    int sel = (lane >> 4) * 16;
    const __bf16* p = A + (size_t)m * lda + k0 + sel;
    v8bf lo = *(const v8bf*)(p);
    v8bf hi = *(const v8bf*)(p + 8);
    return __builtin_shufflevector(lo, hi, 0, 1, 2, 3, 4, 5, 6, 7,
                                   8, 9, 10, 11, 12, 13, 14, 15);
}

// each wave issues 2 async-load instructions per tile (ASYNCcnt += 2)
__device__ inline void async_copy_btile(const __bf16* __restrict__ W, int ldb, int N,
                                        int blockN, int k0, unsigned ldsBase, int tid) {
#pragma unroll
    for (int c = 0; c < 2; ++c) {
        int chunk = tid + c * 256;              // 0..511: 16B pieces of 8KB tile
        int row   = chunk >> 2;                 // 0..127 (N within tile)
        int piece = chunk & 3;                  // 16B piece within 64B row
        int gRow  = blockN + row;
        if (gRow >= N) gRow = N - 1;            // clamp; dup rows never stored
        const __bf16* src = W + (size_t)gRow * ldb + k0 + piece * 8;
        unsigned dst = ldsBase + (unsigned)(row * 64 + piece * 16);
        asm volatile("global_load_async_to_lds_b128 %0, %1, off"
                     :: "v"(dst), "v"(src) : "memory");
    }
}

__global__ __launch_bounds__(256)
void k_gemm_bf16(const __bf16* __restrict__ A, int lda,
                 const __bf16* __restrict__ W, int ldb,
                 float* __restrict__ C, int ldc,
                 int M, int N, int K) {
    __shared__ v16bf smemB[2][128][2];          // 2 x 8KB B tiles
    int tid  = threadIdx.x;
    int lane = tid & 31;
    int wave = tid >> 5;
    int mw = wave & 3, nw = wave >> 2;
    int blockN = blockIdx.x * 128;
    int mBase = blockIdx.y * 128 + mw * 32;
    int nBase = blockN + nw * 64;

    unsigned ldsB0 = (unsigned)(uintptr_t)&smemB[0][0][0];
    unsigned ldsB1 = (unsigned)(uintptr_t)&smemB[1][0][0];

    v8f acc[2][4];
#pragma unroll
    for (int i = 0; i < 2; ++i)
#pragma unroll
        for (int j = 0; j < 4; ++j)
            acc[i][j] = (v8f){0.f, 0.f, 0.f, 0.f, 0.f, 0.f, 0.f, 0.f};

    int nRow = nw * 64 + (lane & 15);
    int sel  = lane >> 4;

    auto compute = [&](int k0, int cur) {
        v16bf a0 = load_a_frag(A, lda, mBase,      k0, lane);
        v16bf a1 = load_a_frag(A, lda, mBase + 16, k0, lane);
        v16bf bfr[4];
#pragma unroll
        for (int j = 0; j < 4; ++j)
            bfr[j] = smemB[cur][nRow + j * 16][sel];
#pragma unroll
        for (int j = 0; j < 4; ++j)
            acc[0][j] = __builtin_amdgcn_wmma_f32_16x16x32_bf16(
                false, a0, false, bfr[j], (short)0, acc[0][j], false, false);
#pragma unroll
        for (int j = 0; j < 4; ++j)
            acc[1][j] = __builtin_amdgcn_wmma_f32_16x16x32_bf16(
                false, a1, false, bfr[j], (short)0, acc[1][j], false, false);
    };

    int nk = K >> 5;
    async_copy_btile(W, ldb, N, blockN, 0, ldsB0, tid);

    // branch-free steady state: prefetch tile kt+1 while computing tile kt
    for (int kt = 0; kt < nk - 1; ++kt) {
        async_copy_btile(W, ldb, N, blockN, (kt + 1) << 5,
                         ((kt + 1) & 1) ? ldsB1 : ldsB0, tid);
        asm volatile("s_wait_asynccnt 0x2" ::: "memory");  // tile kt landed
        __syncthreads();
        compute(kt << 5, kt & 1);
        __syncthreads();                         // reads done before overwrite
    }

    // peeled last iteration: drain all copies, no trailing barrier needed
    asm volatile("s_wait_asynccnt 0x0" ::: "memory");
    __syncthreads();
    compute((nk - 1) << 5, (nk - 1) & 1);

    // store: M always a multiple of 128 here -> no row guard
    int nIn = lane & 15;
    int mRow = (lane >> 4) * 8;
#pragma unroll
    for (int i = 0; i < 2; ++i)
#pragma unroll
        for (int j = 0; j < 4; ++j) {
            int col = nBase + j * 16 + nIn;
            if (col < N) {
                float* cp = C + (size_t)(mBase + i * 16 + mRow) * ldc + col;
#pragma unroll
                for (int r = 0; r < 8; ++r)
                    cp[(size_t)r * ldc] = acc[i][j][r];
            }
        }
}

// ---------------------------------------------------------------------------
// depthwise causal conv1d (width 4) over x = xz[:, 0:DINNER], + SiLU
// ---------------------------------------------------------------------------
__global__ void k_conv_silu(const float* __restrict__ xz, const float* __restrict__ cw,
                            const float* __restrict__ cb, float* __restrict__ xc,
                            __bf16* __restrict__ xcb) {
    int idx = blockIdx.x * blockDim.x + threadIdx.x;   // < MROWS*DINNER
    int d = idx & (DINNER - 1);
    int m = idx >> 11;
    int l = m & (SEQLEN - 1);
    float acc = cb[d];
#pragma unroll
    for (int kk = 0; kk < DCONV; ++kk) {
        int lk = l - (DCONV - 1) + kk;
        if (lk >= 0)
            acc = fmaf(xz[(size_t)(m - (DCONV - 1) + kk) * (2 * DINNER) + d],
                       cw[d * DCONV + kk], acc);
    }
    float o = acc / (1.f + __expf(-acc));              // silu
    xc[idx]  = o;
    xcb[idx] = (__bf16)o;
}

// extract dt slice xdbl[:, 0:DTRANK] as bf16 for the dt_proj GEMM
__global__ void k_cvt_dt(const float* __restrict__ xdbl, __bf16* __restrict__ dtb) {
    int idx = blockIdx.x * blockDim.x + threadIdx.x;   // < MROWS*DTRANK
    int r = idx >> 6, c = idx & (DTRANK - 1);
    dtb[idx] = (__bf16)xdbl[(size_t)r * XDBLW + c];
}

// delta = softplus(delta_raw + dt_bias)
__global__ void k_softplus(float* __restrict__ delta, const float* __restrict__ dtb) {
    int idx = blockIdx.x * blockDim.x + threadIdx.x;   // < MROWS*DINNER
    int d = idx & (DINNER - 1);
    float v = delta[idx] + dtb[d];
    delta[idx] = (v > 20.f) ? v : log1pf(__expf(v));
}

// ---------------------------------------------------------------------------
// selective scan: serial over L, parallel over (b, d) channels.
// lane s = tid&15 handles one of 16 states; 16 channels/block; y_t via
// __shfl_xor reductions within 16-lane groups.
// ---------------------------------------------------------------------------
__global__ void k_scan(const float* __restrict__ delta, const float* __restrict__ u,
                       const float* __restrict__ xdbl, const float* __restrict__ A_log,
                       float* __restrict__ y) {
    int t = threadIdx.x;
    int s = t & (DSTATE - 1);
    int ch = blockIdx.x * 16 + (t >> 4);   // 0..MROWS-1 over (b,d)
    int b = ch >> 11;                      // / DINNER
    int d = ch & (DINNER - 1);
    float A = -__expf(A_log[d * DSTATE + s]);
    float st = 0.f;
    size_t rbase = (size_t)b * SEQLEN;
    for (int l = 0; l < SEQLEN; ++l) {
        size_t m = rbase + l;
        float dv = delta[m * DINNER + d];
        float uv = u[m * DINNER + d];
        float Bt = xdbl[m * XDBLW + DTRANK + s];
        float Ct = xdbl[m * XDBLW + DTRANK + DSTATE + s];
        float dA = __expf(dv * A);
        st = fmaf(dA, st, dv * uv * Bt);
        float part = st * Ct;
        part += __shfl_xor(part, 8, 16);
        part += __shfl_xor(part, 4, 16);
        part += __shfl_xor(part, 2, 16);
        part += __shfl_xor(part, 1, 16);
        if (s == 0) y[m * DINNER + d] = part;
    }
}

// y = (y + xconv * D) * silu(z);  z = xz[:, DINNER + d]
__global__ void k_gate(float* __restrict__ y, const float* __restrict__ xc,
                       const float* __restrict__ Dp, const float* __restrict__ xz,
                       __bf16* __restrict__ yb) {
    int idx = blockIdx.x * blockDim.x + threadIdx.x;   // < MROWS*DINNER
    int d = idx & (DINNER - 1);
    int m = idx >> 11;
    float z = xz[(size_t)m * (2 * DINNER) + DINNER + d];
    float sz = z / (1.f + __expf(-z));
    float v = (y[idx] + xc[idx] * Dp[d]) * sz;
    y[idx]  = v;
    yb[idx] = (__bf16)v;
}

// ---------------------------------------------------------------------------
extern "C" void kernel_launch(void* const* d_in, const int* in_sizes, int n_in,
                              void* d_out, int out_size, void* d_ws, size_t ws_size,
                              hipStream_t stream) {
    const float* hs_in  = (const float*)d_in[0];
    const float* norm_w = (const float*)d_in[1];
    const float* norm_b = (const float*)d_in[2];
    const float* in_w   = (const float*)d_in[3];
    const float* conv_w = (const float*)d_in[4];
    const float* conv_b = (const float*)d_in[5];
    const float* x_w    = (const float*)d_in[6];
    const float* dt_w   = (const float*)d_in[7];
    const float* dt_b   = (const float*)d_in[8];
    const float* A_log  = (const float*)d_in[9];
    const float* Dp     = (const float*)d_in[10];
    const float* out_w  = (const float*)d_in[11];
    const float* fn_w   = (const float*)d_in[12];
    const float* fn_b   = (const float*)d_in[13];
    float* out = (float*)d_out;

    char* ws = (char*)d_ws;
    size_t off = 0;
    auto alloc = [&](size_t bytes) -> void* {
        void* p = ws + off;
        off += (bytes + 255) & ~(size_t)255;
        return p;
    };

    float*  resid   = (float*) alloc((size_t)MROWS * DMODEL * 4);
    float*  hs      = (float*) alloc((size_t)MROWS * DMODEL * 4);
    __bf16* h_bf    = (__bf16*)alloc((size_t)MROWS * DMODEL * 2);
    float*  xz      = (float*) alloc((size_t)MROWS * 2 * DINNER * 4);
    float*  xconv   = (float*) alloc((size_t)MROWS * DINNER * 4);
    __bf16* xconv_b = (__bf16*)alloc((size_t)MROWS * DINNER * 2);
    float*  xdbl    = (float*) alloc((size_t)MROWS * XDBLW * 4);
    __bf16* dt_bf   = (__bf16*)alloc((size_t)MROWS * DTRANK * 2);
    float*  delta   = (float*) alloc((size_t)MROWS * DINNER * 4);
    float*  ybuf    = (float*) alloc((size_t)MROWS * DINNER * 4);
    __bf16* y_bf    = (__bf16*)alloc((size_t)MROWS * DINNER * 2);
    __bf16* inw_bf  = (__bf16*)alloc((size_t)NLAYER * 2 * DINNER * DMODEL * 2);
    __bf16* xw_bf   = (__bf16*)alloc((size_t)NLAYER * XDBLW * DINNER * 2);
    __bf16* dtw_bf  = (__bf16*)alloc((size_t)NLAYER * DINNER * DTRANK * 2);
    __bf16* outw_bf = (__bf16*)alloc((size_t)NLAYER * DMODEL * DINNER * 2);

    // per-launch weight conversion to bf16
    k_cvt_bf16<<<8192, 256, 0, stream>>>(in_w,  inw_bf,  (long)NLAYER * 2 * DINNER * DMODEL);
    k_cvt_bf16<<<1024, 256, 0, stream>>>(x_w,   xw_bf,   (long)NLAYER * XDBLW * DINNER);
    k_cvt_bf16<<<1024, 256, 0, stream>>>(dt_w,  dtw_bf,  (long)NLAYER * DINNER * DTRANK);
    k_cvt_bf16<<<8192, 256, 0, stream>>>(out_w, outw_bf, (long)NLAYER * DMODEL * DINNER);

    // residual starts as the input hidden_states
    hipMemcpyAsync(resid, hs_in, (size_t)MROWS * DMODEL * 4,
                   hipMemcpyDeviceToDevice, stream);

    auto gemm = [&](const __bf16* A, int lda, const __bf16* W, int ldb,
                    float* C, int ldc, int M, int N, int K) {
        dim3 g((N + 127) / 128, (M + 127) / 128);
        k_gemm_bf16<<<g, 256, 0, stream>>>(A, lda, W, ldb, C, ldc, M, N, K);
    };

    const int nElemDI = MROWS * DINNER;
    for (int i = 0; i < NLAYER; ++i) {
        k_add_ln<<<MROWS, 256, 0, stream>>>(
            resid, (i == 0) ? (const float*)nullptr : hs,
            norm_w + (size_t)i * DMODEL, norm_b + (size_t)i * DMODEL,
            h_bf, nullptr);

        // in_proj: xz[M,4096] = h[M,1024] * in_w[4096,1024]^T
        gemm(h_bf, DMODEL, inw_bf + (size_t)i * 2 * DINNER * DMODEL, DMODEL,
             xz, 2 * DINNER, MROWS, 2 * DINNER, DMODEL);

        k_conv_silu<<<nElemDI / 256, 256, 0, stream>>>(
            xz, conv_w + (size_t)i * DINNER * DCONV, conv_b + (size_t)i * DINNER,
            xconv, xconv_b);

        // x_proj: xdbl[M,96] = xconv[M,2048] * x_w[96,2048]^T
        gemm(xconv_b, DINNER, xw_bf + (size_t)i * XDBLW * DINNER, DINNER,
             xdbl, XDBLW, MROWS, XDBLW, DINNER);

        // dt_proj: delta_raw[M,2048] = dt[M,64] * dt_w[2048,64]^T
        k_cvt_dt<<<MROWS * DTRANK / 256, 256, 0, stream>>>(xdbl, dt_bf);
        gemm(dt_bf, DTRANK, dtw_bf + (size_t)i * DINNER * DTRANK, DTRANK,
             delta, DINNER, MROWS, DINNER, DTRANK);
        k_softplus<<<nElemDI / 256, 256, 0, stream>>>(delta, dt_b + (size_t)i * DINNER);

        // selective scan
        k_scan<<<MROWS / 16, 256, 0, stream>>>(
            delta, xconv, xdbl, A_log + (size_t)i * DINNER * DSTATE, ybuf);

        k_gate<<<nElemDI / 256, 256, 0, stream>>>(
            ybuf, xconv, Dp + (size_t)i * DINNER, xz, y_bf);

        // out_proj: hs[M,1024] = y[M,2048] * out_w[1024,2048]^T
        gemm(y_bf, DINNER, outw_bf + (size_t)i * DMODEL * DINNER, DINNER,
             hs, DMODEL, MROWS, DMODEL, DINNER);
    }

    k_add_ln<<<MROWS, 256, 0, stream>>>(resid, hs, fn_w, fn_b, nullptr, out);
}
